// GPTAttention_40931038331105
// MI455X (gfx1250) — compile-verified, matrix operand order checked by reading, not run on hardware
//
#include <hip/hip_runtime.h>
#include <hip/hip_bf16.h>

#define BB    2
#define TT    2048
#define CC    1024
#define NHEAD 16
#define HD    64
#define N3C   3072
#define MROWS (BB*TT)   // 4096

typedef __bf16 bf16;
typedef __attribute__((ext_vector_type(16))) __bf16 bf16x16;
typedef __attribute__((ext_vector_type(8)))  __bf16 bf16x8;
typedef __attribute__((ext_vector_type(8)))  float  f32x8;
typedef __attribute__((ext_vector_type(4)))  int    i32x4;

static __device__ __forceinline__ bf16x16 cat8(bf16x8 lo, bf16x8 hi) {
  bf16x16 r;
#pragma unroll
  for (int i = 0; i < 8; ++i) { r[i] = lo[i]; r[i + 8] = hi[i]; }
  return r;
}
static __device__ __forceinline__ bf16x8 ldb8(const bf16* p) {
  return *reinterpret_cast<const bf16x8*>(p);
}
static __device__ __forceinline__ f32x8 wmma_bf16(bf16x16 a, bf16x16 b, f32x8 c) {
  return __builtin_amdgcn_wmma_f32_16x16x32_bf16(false, a, false, b, (short)0, c, false, false);
}

// ---- CDNA5 async global->LDS copy (16B per lane), ASYNCcnt tracked ------------
typedef __attribute__((address_space(1))) i32x4 as1_i32x4;
typedef __attribute__((address_space(3))) i32x4 as3_i32x4;

static __device__ __forceinline__ void async_copy16(const bf16* g, bf16* l) {
#if __has_builtin(__builtin_amdgcn_global_load_async_to_lds_b128)
  __builtin_amdgcn_global_load_async_to_lds_b128(
      (as1_i32x4*)(uintptr_t)g,
      (as3_i32x4*)(unsigned)(uintptr_t)l,   // low 32 bits of generic LDS ptr = LDS offset
      0, 0);
#else
  unsigned loff = (unsigned)(uintptr_t)l;
  asm volatile("global_load_async_to_lds_b128 %0, %1, off"
               :: "v"(loff), "v"(g) : "memory");
#endif
}
static __device__ __forceinline__ void wait_async0() {
#if __has_builtin(__builtin_amdgcn_s_wait_asynccnt)
  __builtin_amdgcn_s_wait_asynccnt(0);
#else
  asm volatile("s_wait_asynccnt 0" ::: "memory");
#endif
}
static __device__ __forceinline__ void wait_async16() {
#if __has_builtin(__builtin_amdgcn_s_wait_asynccnt)
  __builtin_amdgcn_s_wait_asynccnt(16);
#else
  asm volatile("s_wait_asynccnt 16" ::: "memory");
#endif
}

// ---------------- conversion kernels ----------------
__global__ void cvt_x_kernel(const float* __restrict__ x, bf16* __restrict__ xbf, int n) {
  int i = blockIdx.x * 256 + threadIdx.x;
  if (i < n) xbf[i] = (bf16)x[i];
}

// wT[n][k] = (bf16) w[k][n]   (w is [K][N] row-major)
__global__ void cvt_transpose_kernel(const float* __restrict__ w, bf16* __restrict__ wT,
                                     int K, int N) {
  int i = blockIdx.x * 256 + threadIdx.x;
  if (i < K * N) {
    int n = i / K, k = i - n * K;
    wT[i] = (bf16)w[(size_t)k * N + n];
  }
}

// ======== shared GEMM core: block = 4 waves, 64Mx64N tile, LDS double-buffer ====
#define LROW 72   // 64 + 8 pad elements (144B rows)

// Issues one 64x64 bf16 tile (64 rows x 8 chunks of 16B) as async copies.
static __device__ __forceinline__ void stage_tile_async(
    const bf16* __restrict__ gbase, int row0, int k0, int ldg,
    bf16 (*lds)[LROW], int tid) {
#pragma unroll
  for (int i = 0; i < 4; ++i) {
    int c = tid + 128 * i;            // 0..511
    int row = c >> 3;
    int col = (c & 7) * 8;            // element offset, 16B chunks
    async_copy16(gbase + (size_t)(row0 + row) * ldg + k0 + col, &lds[row][col]);
  }
}

// Computes one stage (K=64, two WMMA k-substeps) for this wave's 16Mx64N strip.
static __device__ __forceinline__ void compute_stage(
    const bf16 (*As)[LROW], const bf16 (*Bs)[LROW],
    int wid, int l16, int koff, f32x8 acc[4]) {
#pragma unroll
  for (int s = 0; s < 2; ++s) {
    const bf16* arow = &As[wid * 16 + l16][s * 32 + koff];
    bf16x16 a = cat8(ldb8(arow), ldb8(arow + 16));
    bf16x16 bm[4];
#pragma unroll
    for (int nt = 0; nt < 4; ++nt) {
      const bf16* brow = &Bs[nt * 16 + l16][s * 32 + koff];
      bm[nt] = cat8(ldb8(brow), ldb8(brow + 16));
    }
#pragma unroll
    for (int nt = 0; nt < 4; ++nt) acc[nt] = wmma_bf16(a, bm[nt], acc[nt]);
  }
}

// ---------------- QKV GEMM: [4096,1024] x [1024,3072] + bias -> Q,K,Vt (bf16) ----
__global__ void __launch_bounds__(128)
qkv_gemm_kernel(const bf16* __restrict__ xbf,   // [4096][1024]
                const bf16* __restrict__ wT,    // [3072][1024]
                const float* __restrict__ b_attn,
                bf16* __restrict__ q,           // [B*H][T][64]
                bf16* __restrict__ k,           // [B*H][T][64]
                bf16* __restrict__ vT) {        // [B*H][64][T]
  __shared__ __align__(16) bf16 As[2][64][LROW];
  __shared__ __align__(16) bf16 Bs[2][64][LROW];
  const int tid  = threadIdx.x;
  const int lane = tid & 31;
  const int wid  = tid >> 5;
  const int l16  = lane & 15;
  const int koff = (lane >> 4) * 8;           // 0 or 8
  const int mblk = blockIdx.y * 64;
  const int n0   = blockIdx.x * 64;

  f32x8 acc[4] = {};
  stage_tile_async(xbf, mblk, 0, CC, As[0], tid);
  stage_tile_async(wT,  n0,   0, CC, Bs[0], tid);
  const int NS = CC / 64;                     // 16 stages
  for (int ks = 0; ks < NS; ++ks) {
    wait_async0();
    __syncthreads();
    if (ks + 1 < NS) {
      stage_tile_async(xbf, mblk, (ks + 1) * 64, CC, As[(ks + 1) & 1], tid);
      stage_tile_async(wT,  n0,   (ks + 1) * 64, CC, Bs[(ks + 1) & 1], tid);
    }
    compute_stage(As[ks & 1], Bs[ks & 1], wid, l16, koff, acc);
    __syncthreads();
  }

  const int m0 = mblk + wid * 16;
#pragma unroll
  for (int nt = 0; nt < 4; ++nt) {
    const int ncol = n0 + nt * 16 + l16;
    const float bias = b_attn[ncol];
    const int sel = ncol >> 10;        // 0=q 1=k 2=v
    const int nc  = ncol & 1023;
    const int h = nc >> 6, d = nc & 63;
#pragma unroll
    for (int r = 0; r < 8; ++r) {
      const int m = m0 + r + koff;     // koff==8 -> upper half rows
      const int b = m >> 11, t = m & 2047;
      const size_t bh = (size_t)(b * NHEAD + h);
      const float val = acc[nt][r] + bias;
      if (sel == 0)      q [(bh * TT + t) * HD + d] = (bf16)val;
      else if (sel == 1) k [(bh * TT + t) * HD + d] = (bf16)val;
      else               vT[(bh * HD + d) * TT + t] = (bf16)val;
    }
  }
}

// ---------------- Flash attention: one wave per (b,h,16-query tile) -------------
// K/V chunks (32 keys) double-buffered in LDS via async copies, per wave.
#define KROW 72   // K tile row: 64 d + pad
#define VROW 40   // V tile row: 32 keys + pad

// Stage one 32-key chunk of K (32x64) and Vt (64x32) as 16 async copies per lane.
static __device__ __forceinline__ void stage_kv_async(
    const bf16* __restrict__ kg,   // kmat + (bh*T + k0)*HD
    const bf16* __restrict__ vg,   // vT + bh*HD*T + k0
    bf16 (*kl)[KROW], bf16 (*vl)[VROW], int lane) {
#pragma unroll
  for (int i = 0; i < 8; ++i) {
    int c = lane + 32 * i;           // 0..255
    int kr = c >> 3, kc = (c & 7) * 8;
    async_copy16(kg + (size_t)kr * HD + kc, &kl[kr][kc]);
  }
#pragma unroll
  for (int i = 0; i < 8; ++i) {
    int c = lane + 32 * i;           // 0..255
    int vr = c >> 2, vc = (c & 3) * 8;
    async_copy16(vg + (size_t)vr * TT + vc, &vl[vr][vc]);
  }
}

__global__ void __launch_bounds__(128)
attn_kernel(const bf16* __restrict__ q,
            const bf16* __restrict__ kmat,
            const bf16* __restrict__ vT,
            bf16* __restrict__ ybf) {           // [B][T][C]
  __shared__ __align__(16) bf16 Kst[4][2][32][KROW];
  __shared__ __align__(16) bf16 Vst[4][2][64][VROW];
  __shared__ __align__(16) bf16 Plds[4][16][32];
  const int lane = threadIdx.x & 31;
  const int wid  = threadIdx.x >> 5;
  const int l16  = lane & 15;
  const int half = lane >> 4;
  const int koff = half * 8;
  const int g  = blockIdx.x * 4 + wid;
  const int bh = g >> 7;                 // 0..31
  const int q0 = (g & 127) * 16;

  const bf16* qbase = q + ((size_t)bh * TT + q0 + l16) * HD;
  bf16x16 qa[2];
  qa[0] = cat8(ldb8(qbase + koff),      ldb8(qbase + koff + 16));
  qa[1] = cat8(ldb8(qbase + 32 + koff), ldb8(qbase + 32 + koff + 16));

  f32x8 o[4] = {};
  float mrow[8], lrow[8];
#pragma unroll
  for (int r = 0; r < 8; ++r) { mrow[r] = -3.0e38f; lrow[r] = 0.0f; }
  const float scale = 0.125f;            // 1/sqrt(64)

  const bf16* kbase = kmat + (size_t)bh * TT * HD;
  const bf16* vbase = vT + (size_t)bh * HD * TT;
  const int kend = q0 + 16;

  stage_kv_async(kbase, vbase, Kst[wid][0], Vst[wid][0], lane);
  int buf = 0;
  for (int k0 = 0; k0 < kend; k0 += 32) {
    const bool nxt = (k0 + 32) < kend;
    if (nxt) {
      asm volatile("s_wait_dscnt 0" ::: "memory");   // older buffer's LDS reads done
      stage_kv_async(kbase + (size_t)(k0 + 32) * HD, vbase + (k0 + 32),
                     Kst[wid][buf ^ 1], Vst[wid][buf ^ 1], lane);
      wait_async16();                                // current chunk's 16 copies landed
    } else {
      wait_async0();
    }

    f32x8 s[2] = {};
#pragma unroll
    for (int kt = 0; kt < 2; ++kt) {
      const bf16* kb = &Kst[wid][buf][kt * 16 + l16][0];
      bf16x16 b0 = cat8(ldb8(kb + koff),      ldb8(kb + koff + 16));
      bf16x16 b1 = cat8(ldb8(kb + 32 + koff), ldb8(kb + 32 + koff + 16));
      s[kt] = wmma_bf16(qa[0], b0, s[kt]);
      s[kt] = wmma_bf16(qa[1], b1, s[kt]);
    }
    // masked online softmax; lanes 0-15 own rows 0-7, lanes 16-31 own rows 8-15
#pragma unroll
    for (int r = 0; r < 8; ++r) {
      const int qrow = q0 + half * 8 + r;
      const float v0 = ((k0 + l16)      <= qrow) ? s[0][r] * scale : -3.0e38f;
      const float v1 = ((k0 + 16 + l16) <= qrow) ? s[1][r] * scale : -3.0e38f;
      float mx = fmaxf(v0, v1);
#pragma unroll
      for (int off = 8; off >= 1; off >>= 1) mx = fmaxf(mx, __shfl_xor(mx, off, 32));
      const float mn = fmaxf(mrow[r], mx);
      const float p0 = __expf(v0 - mn);
      const float p1 = __expf(v1 - mn);
      float sum = p0 + p1;
#pragma unroll
      for (int off = 8; off >= 1; off >>= 1) sum += __shfl_xor(sum, off, 32);
      const float alpha = __expf(mrow[r] - mn);
      lrow[r] = lrow[r] * alpha + sum;
      mrow[r] = mn;
#pragma unroll
      for (int dt = 0; dt < 4; ++dt) o[dt][r] *= alpha;
      Plds[wid][half * 8 + r][l16]      = (bf16)p0;
      Plds[wid][half * 8 + r][16 + l16] = (bf16)p1;
    }
    asm volatile("s_wait_dscnt 0" ::: "memory");
    bf16x16 pa = cat8(ldb8(&Plds[wid][l16][koff]), ldb8(&Plds[wid][l16][koff + 16]));
#pragma unroll
    for (int dt = 0; dt < 4; ++dt) {
      const bf16* vb = &Vst[wid][buf][dt * 16 + l16][0];
      bf16x16 bm = cat8(ldb8(vb + koff), ldb8(vb + koff + 16));
      o[dt] = wmma_bf16(pa, bm, o[dt]);
    }
    buf ^= 1;
  }
  const int h = bh & 15, b = bh >> 4;
#pragma unroll
  for (int r = 0; r < 8; ++r) {
    const float inv = 1.0f / lrow[r];
    const int t = q0 + half * 8 + r;
#pragma unroll
    for (int dt = 0; dt < 4; ++dt) {
      const int c = h * HD + dt * 16 + l16;
      ybf[((size_t)b * TT + t) * CC + c] = (bf16)(o[dt][r] * inv);
    }
  }
}

// ---------------- Proj GEMM: [4096,1024] x [1024,1024] + bias -> f32 out --------
__global__ void __launch_bounds__(128)
proj_gemm_kernel(const bf16* __restrict__ ybf,  // [4096][1024]
                 const bf16* __restrict__ wT,   // [1024][1024]
                 const float* __restrict__ b_proj,
                 float* __restrict__ out) {     // [4096][1024]
  __shared__ __align__(16) bf16 As[2][64][LROW];
  __shared__ __align__(16) bf16 Bs[2][64][LROW];
  const int tid  = threadIdx.x;
  const int lane = tid & 31;
  const int wid  = tid >> 5;
  const int l16  = lane & 15;
  const int koff = (lane >> 4) * 8;
  const int mblk = blockIdx.y * 64;
  const int n0   = blockIdx.x * 64;

  f32x8 acc[4] = {};
  stage_tile_async(ybf, mblk, 0, CC, As[0], tid);
  stage_tile_async(wT,  n0,   0, CC, Bs[0], tid);
  const int NS = CC / 64;
  for (int ks = 0; ks < NS; ++ks) {
    wait_async0();
    __syncthreads();
    if (ks + 1 < NS) {
      stage_tile_async(ybf, mblk, (ks + 1) * 64, CC, As[(ks + 1) & 1], tid);
      stage_tile_async(wT,  n0,   (ks + 1) * 64, CC, Bs[(ks + 1) & 1], tid);
    }
    compute_stage(As[ks & 1], Bs[ks & 1], wid, l16, koff, acc);
    __syncthreads();
  }

  const int m0 = mblk + wid * 16;
#pragma unroll
  for (int nt = 0; nt < 4; ++nt) {
    const int ncol = n0 + nt * 16 + l16;
    const float bias = b_proj[ncol];
#pragma unroll
    for (int r = 0; r < 8; ++r) {
      const int m = m0 + r + koff;
      out[(size_t)m * CC + ncol] = acc[nt][r] + bias;
    }
  }
}

// ---------------- launch --------------------------------------------------------
extern "C" void kernel_launch(void* const* d_in, const int* in_sizes, int n_in,
                              void* d_out, int out_size, void* d_ws, size_t ws_size,
                              hipStream_t stream) {
  const float* x      = (const float*)d_in[0];   // [2,2048,1024]
  const float* w_attn = (const float*)d_in[1];   // [1024,3072]
  const float* b_attn = (const float*)d_in[2];   // [3072]
  const float* w_proj = (const float*)d_in[3];   // [1024,1024]
  const float* b_proj = (const float*)d_in[4];   // [1024]
  float* out = (float*)d_out;

  char* ws = (char*)d_ws;
  const size_t szXBF = (size_t)MROWS * CC * 2;   // 8 MiB
  const size_t szWAT = (size_t)N3C * CC * 2;     // 6 MiB
  const size_t szWPT = (size_t)CC * CC * 2;      // 2 MiB
  const size_t szQ   = (size_t)MROWS * CC * 2;   // 8 MiB (each of q,k,vT,y)
  bf16* xbf    = (bf16*)(ws);
  bf16* wattnT = (bf16*)(ws + szXBF);
  bf16* wprojT = (bf16*)(ws + szXBF + szWAT);
  bf16* qbuf   = (bf16*)(ws + szXBF + szWAT + szWPT);
  bf16* kbuf   = (bf16*)(ws + szXBF + szWAT + szWPT + szQ);
  bf16* vTbuf  = (bf16*)(ws + szXBF + szWAT + szWPT + 2 * szQ);
  bf16* ybuf   = (bf16*)(ws + szXBF + szWAT + szWPT + 3 * szQ);

  cvt_x_kernel<<<(MROWS * CC + 255) / 256, 256, 0, stream>>>(x, xbf, MROWS * CC);
  cvt_transpose_kernel<<<(N3C * CC + 255) / 256, 256, 0, stream>>>(w_attn, wattnT, CC, N3C);
  cvt_transpose_kernel<<<(CC * CC + 255) / 256, 256, 0, stream>>>(w_proj, wprojT, CC, CC);

  qkv_gemm_kernel<<<dim3(N3C / 64, MROWS / 64), 128, 0, stream>>>(
      xbf, wattnT, b_attn, qbuf, kbuf, vTbuf);

  attn_kernel<<<(32 * 128) / 4, 128, 0, stream>>>(qbuf, kbuf, vTbuf, ybuf);

  proj_gemm_kernel<<<dim3(CC / 64, MROWS / 64), 128, 0, stream>>>(
      ybuf, wprojT, b_proj, out);
}
